// SpatialAttention_80453327389435
// MI455X (gfx1250) — matrix operand order repaired and checked by reading, not compiled, not run on hardware
//
#include <hip/hip_runtime.h>
#include <hip/hip_bf16.h>

typedef __attribute__((ext_vector_type(16))) __bf16 v16bf;
typedef __attribute__((ext_vector_type(8)))  __bf16 v8bf;
typedef __attribute__((ext_vector_type(8)))  float  v8f;

#define LDX 264   // row stride (bf16) for X slices (K=256) and Wt (d=256); 528B = 33*16
#define LDQ 72    // row stride for Q,K tiles (e=64); 144B = 9*16
#define LDV 136   // row stride for Vt (b=128) and P (b=128); 272B = 17*16

struct __align__(16) Smem {
  __bf16 Xq[128 * LDX];
  __bf16 Xk[128 * LDX];
  __bf16 Xv[128 * LDX];
  __bf16 Q [128 * LDQ];   // Q[m][e], A-operand of Q*K^T
  __bf16 K [128 * LDQ];   // K[b][e], used as B^T (col-major B) of Q*K^T
  __bf16 Vt[64  * LDV];   // V^T[e][b] (pre-scaled by w_head[h]), B^T of P*V
  union {
    __bf16 Wt[64 * LDX];  // W^T[e][d], B^T of projections (alive only during proj phase)
    __bf16 P [128 * LDV]; // softmax(S)[m][b], A-operand of P*V (alive only after proj phase)
  } u;
};

__device__ __forceinline__ v8f wmma_bf16(v16bf a, v16bf b, v8f c) {
  // D = A(16x32) * B(32x16) + C, f32 accumulate
  return __builtin_amdgcn_wmma_f32_16x16x32_bf16(false, a, false, b, (short)0, c,
                                                 false, false);
}

// A fragment (16x32 bf16): lane L -> row m0+(L&15); VGPR j holds
// k = (j>>2)*16 + (L>>4)*8 + (j&3)*2 (+0/1)  => two contiguous 16B chunks per lane.
__device__ __forceinline__ v16bf load_a(const __bf16* base, int ld, int m0, int kb) {
  const int lane = threadIdx.x & 31;
  const int hi   = lane >> 4;
  const __bf16* p = base + (m0 + (lane & 15)) * ld + kb + hi * 8;
  v8bf lo = *(const v8bf*)(p);
  v8bf hv = *(const v8bf*)(p + 16);
  v16bf r;
#pragma unroll
  for (int i = 0; i < 8; ++i) { r[i] = lo[i]; r[8 + i] = hv[i]; }
  return r;
}

// B fragment (32x16 bf16), source stored as B^T (row n, contiguous k):
// lane L -> col n0+(L&15); VGPR j holds k = (j>>2)*16 + (L>>4)*8 + (j&3)*2 (+0/1) => two 16B chunks.
__device__ __forceinline__ v16bf load_b(const __bf16* base, int ld, int n0, int kb) {
  const int lane = threadIdx.x & 31;
  const int hi   = lane >> 4;
  const __bf16* p = base + (n0 + (lane & 15)) * ld + kb + hi * 8;
  v8bf lo = *(const v8bf*)(p);
  v8bf hv = *(const v8bf*)(p + 16);
  v16bf r;
#pragma unroll
  for (int i = 0; i < 8; ++i) { r[i] = lo[i]; r[8 + i] = hv[i]; }
  return r;
}

__global__ __launch_bounds__(256, 1)
void spatial_attn_fused(const float* __restrict__ Xqg, const float* __restrict__ Xkg,
                        const float* __restrict__ Xvg, const float* __restrict__ mask,
                        const float* __restrict__ Wq,  const float* __restrict__ Wk,
                        const float* __restrict__ Wv,  const float* __restrict__ whead,
                        float* __restrict__ attn_out,  float* __restrict__ scores_out) {
  __shared__ Smem sm;
  const int bid  = blockIdx.x;       // bid = t*32 + l  (T=16, L=32)
  const int tid  = threadIdx.x;
  const int lane = tid & 31;
  const int wv   = tid >> 5;         // wave 0..7 -> M-tile (rows 16*wv..16*wv+15)
  const int hi   = lane >> 4;
  const int l15  = lane & 15;
  const int m_base = 16 * wv + 8 * hi;   // + j gives the C/D row per ISA layout

  // ---- stage x slices [128 x 256] f32 -> bf16 LDS (once per block) ----
  {
    const float* srcs[3] = {Xqg, Xkg, Xvg};
    __bf16* dsts[3] = {sm.Xq, sm.Xk, sm.Xv};
    const size_t base = (size_t)bid * 256;   // ((b*T+t)*L+l)*256 = b*131072 + bid*256
    for (int p = 0; p < 3; ++p) {
      const float* s = srcs[p];
      __bf16* d = dsts[p];
      for (int i = tid; i < 128 * 256; i += 256) {
        const int b = i >> 8, dd = i & 255;
        d[b * LDX + dd] = (__bf16)s[(size_t)b * 131072 + base + dd];
      }
    }
  }
  __syncthreads();

  v8f o_acc[4] = {};   // attn_output accumulator: rows 16*wv.., e = 0..63

  for (int h = 0; h < 8; ++h) {
    const float wh = whead[h];

    // ---- projections: Q = Xq*Wq_h, K = Xk*Wk_h, V = Xv*Wv_h (each 128x256 * 256x64)
    for (int p = 0; p < 3; ++p) {
      __syncthreads();   // prior readers of u.Wt / u.P / Q / K / Vt are done
      const float* Wsrc = (p == 0) ? Wq : (p == 1) ? Wk : Wv;
      for (int i = tid; i < 256 * 64; i += 256) {   // stage W^T[e][d] as bf16
        const int dd = i >> 6, e = i & 63;
        sm.u.Wt[e * LDX + dd] = (__bf16)Wsrc[(size_t)dd * 512 + h * 64 + e];
      }
      __syncthreads();

      const __bf16* Xl = (p == 0) ? sm.Xq : (p == 1) ? sm.Xk : sm.Xv;
      v8f acc[4] = {};
      for (int kc = 0; kc < 8; ++kc) {             // K = 256 in steps of 32
        const v16bf a = load_a(Xl, LDX, 16 * wv, kc * 32);
#pragma unroll
        for (int nt = 0; nt < 4; ++nt) {
          const v16bf b = load_b(sm.u.Wt, LDX, 16 * nt, kc * 32);
          acc[nt] = wmma_bf16(a, b, acc[nt]);
        }
      }
#pragma unroll
      for (int nt = 0; nt < 4; ++nt)
#pragma unroll
        for (int j = 0; j < 8; ++j) {
          const int m = m_base + j, n = l15 + 16 * nt;
          const float v = acc[nt][j];
          if (p == 0)      sm.Q[m * LDQ + n]  = (__bf16)v;
          else if (p == 1) sm.K[m * LDQ + n]  = (__bf16)v;
          else             sm.Vt[n * LDV + m] = (__bf16)(v * wh); // fold w_head into V
        }
    }
    __syncthreads();   // Q, K, Vt ready

    // ---- S = Q * K^T (128x128), each wave owns a 16-row stripe ----
    v8f s[8] = {};
#pragma unroll
    for (int kc = 0; kc < 2; ++kc) {               // e = 64 in steps of 32
      const v16bf a = load_a(sm.Q, LDQ, 16 * wv, kc * 32);
#pragma unroll
      for (int nt = 0; nt < 8; ++nt) {
        const v16bf b = load_b(sm.K, LDQ, 16 * nt, kc * 32);
        s[nt] = wmma_bf16(a, b, s[nt]);
      }
    }

    // scale (1/sqrt(64)) * mask, emit scores[t,l,h,B,b] (pre-softmax output)
    float* sc = scores_out + (((size_t)bid * 8 + h) << 14);
#pragma unroll
    for (int nt = 0; nt < 8; ++nt)
#pragma unroll
      for (int j = 0; j < 8; ++j) {
        const int m = m_base + j, n = l15 + 16 * nt;
        const float v = s[nt][j] * 0.125f * mask[(m << 7) + n];
        s[nt][j] = v;
        sc[(m << 7) + n] = v;
      }

    // softmax over n (axis b): rows live in (j, lane-half); 16-lane shuffle reduce
#pragma unroll
    for (int j = 0; j < 8; ++j) {
      float mx = s[0][j];
#pragma unroll
      for (int nt = 1; nt < 8; ++nt) mx = fmaxf(mx, s[nt][j]);
#pragma unroll
      for (int d = 1; d < 16; d <<= 1) mx = fmaxf(mx, __shfl_xor(mx, d, 32));
      float sum = 0.f;
#pragma unroll
      for (int nt = 0; nt < 8; ++nt) {
        const float e = __expf(s[nt][j] - mx);
        s[nt][j] = e;
        sum += e;
      }
#pragma unroll
      for (int d = 1; d < 16; d <<= 1) sum += __shfl_xor(sum, d, 32);
      const float inv = 1.f / sum;
      const int m = m_base + j;
#pragma unroll
      for (int nt = 0; nt < 8; ++nt)
        sm.u.P[m * LDV + l15 + 16 * nt] = (__bf16)(s[nt][j] * inv);
    }
    __syncthreads();   // all P rows written (u.Wt no longer live)

    // ---- O += P * (wh * V): 128x128 * 128x64, accumulate across heads ----
#pragma unroll
    for (int kc = 0; kc < 4; ++kc) {               // b = 128 in steps of 32
      const v16bf a = load_a(sm.u.P, LDV, 16 * wv, kc * 32);
#pragma unroll
      for (int nt = 0; nt < 4; ++nt) {
        const v16bf b = load_b(sm.Vt, LDV, 16 * nt, kc * 32);
        o_acc[nt] = wmma_bf16(a, b, o_acc[nt]);
      }
    }
    // loop-top __syncthreads protects u.P / Q / K / Vt reuse next head
  }

  // ---- attn_output[b,t,l,e] = o_acc ----
#pragma unroll
  for (int nt = 0; nt < 4; ++nt)
#pragma unroll
    for (int j = 0; j < 8; ++j) {
      const int m = m_base + j, n = l15 + 16 * nt;
      attn_out[(size_t)m * 32768 + (size_t)bid * 64 + n] = o_acc[nt][j];
    }
}

extern "C" void kernel_launch(void* const* d_in, const int* in_sizes, int n_in,
                              void* d_out, int out_size, void* d_ws, size_t ws_size,
                              hipStream_t stream) {
  (void)in_sizes; (void)n_in; (void)d_ws; (void)ws_size; (void)out_size;
  const float* q    = (const float*)d_in[0];
  const float* k    = (const float*)d_in[1];
  const float* v    = (const float*)d_in[2];
  const float* mask = (const float*)d_in[3];
  const float* Wq   = (const float*)d_in[4];
  const float* Wk   = (const float*)d_in[5];
  const float* Wv   = (const float*)d_in[6];
  const float* wh   = (const float*)d_in[7];
  float* out        = (float*)d_out;
  float* attn_out   = out;                          // [128,16,32,64]
  float* scores_out = out + (size_t)128 * 16 * 32 * 64;  // [16,32,8,128,128]

  spatial_attn_fused<<<dim3(16 * 32), dim3(256), 0, stream>>>(
      q, k, v, mask, Wq, Wk, Wv, wh, attn_out, scores_out);
}
